// DiscreteAttnTRBlock_41231686042368
// MI455X (gfx1250) — matrix operand order repaired and checked by reading, not compiled, not run on hardware
//
#include <hip/hip_runtime.h>

// ---------- types ----------
typedef __attribute__((ext_vector_type(16))) __bf16 v16bf;
typedef __attribute__((ext_vector_type(8)))  __bf16 bf16x8;
typedef __attribute__((ext_vector_type(8)))  float  v8f;
typedef __attribute__((ext_vector_type(4))) unsigned int u32x4;
typedef __attribute__((ext_vector_type(8))) int          i32x8;
typedef __attribute__((ext_vector_type(4))) int          i32x4;

#define CCH 128   // channels

#if __has_builtin(__builtin_amdgcn_tensor_load_to_lds) && \
    __has_builtin(__builtin_amdgcn_s_wait_tensorcnt)
#define USE_TDM 1
#endif

__device__ __forceinline__ void sched_fence() {
#if __has_builtin(__builtin_amdgcn_sched_barrier)
  __builtin_amdgcn_sched_barrier(0);
#endif
}

__device__ __forceinline__ v16bf cat16(bf16x8 lo, bf16x8 hi) {
  return __builtin_shufflevector(lo, hi, 0,1,2,3,4,5,6,7,8,9,10,11,12,13,14,15);
}

// A fragment: 16x32 bf16. lanes 0-15: row M=lane, K = kc*32 + {0..7, 16..23}
//             lanes 16-31: row M=lane-16, K = kc*32 + {8..15, 24..31}
__device__ __forceinline__ v16bf load_a_frag(const __bf16* rowPtr, int kc, int lane) {
  int off = kc * 32 + ((lane & 16) ? 8 : 0);
  bf16x8 lo = *(const bf16x8*)(rowPtr + off);
  bf16x8 hi = *(const bf16x8*)(rowPtr + off + 16);
  return cat16(lo, hi);
}

// B fragment: pre-swizzled, 512 bf16 per fragment, lane's 16 elems contiguous.
__device__ __forceinline__ v16bf load_b_frag(const __bf16* frag, int lane) {
  const __bf16* p = frag + lane * 16;
  bf16x8 lo = *(const bf16x8*)p;
  bf16x8 hi = *(const bf16x8*)(p + 8);
  return cat16(lo, hi);
}

__device__ __forceinline__ v8f wmma_bf16(v16bf a, v16bf b, v8f c) {
  return __builtin_amdgcn_wmma_f32_16x16x32_bf16(false, a, false, b, (short)0, c, false, false);
}

#ifdef USE_TDM
// TDM: DMA one 32KB weight tap (8192 x 4B, 1-D tile) from global into LDS.
__device__ __forceinline__ void tdm_stage(const void* src, unsigned ldsOff) {
  unsigned long long ga = (unsigned long long)src;
  u32x4 g0 = {};
  g0[0] = 1u;                                           // count=1, user mode
  g0[1] = ldsOff;                                       // lds_addr [63:32]
  g0[2] = (unsigned)ga;                                 // global_addr low
  g0[3] = (unsigned)((ga >> 32) & 0x1FFFFFFu) | (2u << 30); // ga hi + type=2
  i32x8 g1 = {};
  g1[0] = (int)(2u << 16);        // data_size = 4 bytes
  g1[1] = (int)(8192u << 16);     // tensor_dim0[15:0] -> bits 63:48
  g1[2] = (int)((8192u >> 16) | (1u << 16)); // dim0 hi, tensor_dim1=1
  g1[3] = (int)(8192u << 16);     // tile_dim0 = 8192 -> bits 127:112
  g1[4] = 0;                      // tile_dim1/2 unused
  g1[5] = 8192;                   // tensor_dim0_stride low32
  g1[6] = 0;
  g1[7] = 0;
  i32x4 g2 = {};
  i32x4 g3 = {};
#if __clang_major__ >= 23
  i32x8 g4 = {};
  __builtin_amdgcn_tensor_load_to_lds(g0, g1, g2, g3, g4, 0);
#else
  __builtin_amdgcn_tensor_load_to_lds(g0, g1, g2, g3, 0);
#endif
}
#endif

// ---------- prep kernels ----------
__global__ void cvt_bf16_kernel(const float* __restrict__ x, __bf16* __restrict__ o, long total) {
  for (long i = (long)blockIdx.x * blockDim.x + threadIdx.x; i < total;
       i += (long)gridDim.x * blockDim.x)
    o[i] = (__bf16)x[i];
}

// swizzle W[taps][Cin][Cout] f32 -> fragment layout bf16
__global__ void swz_kernel(const float* __restrict__ W, __bf16* __restrict__ o,
                           int taps, int Cin, int Cout) {
  const int NT = Cout >> 4, KC = Cin >> 5;
  long total = (long)taps * Cin * Cout;
  for (long i = (long)blockIdx.x * blockDim.x + threadIdx.x; i < total;
       i += (long)gridDim.x * blockDim.x) {
    long t = i;
    int e    = (int)(t & 15); t >>= 4;
    int lane = (int)(t & 31); t >>= 5;
    int nt   = (int)(t % NT); t /= NT;
    int kc   = (int)(t % KC);
    int tap  = (int)(t / KC);
    int K = kc * 32 + ((e >> 3) << 4) + ((lane >> 4) << 3) + (e & 7);
    int Nc = nt * 16 + (lane & 15);
    o[i] = (__bf16)W[((long)tap * Cin + K) * Cout + Nc];
  }
}

// ---------- sparse conv (27-tap): WMMA, 32 rows/wave, TDM double-buffered B ----------
__global__ __launch_bounds__(256)
void spconv_kernel(const __bf16* __restrict__ xb, const int* __restrict__ nbr,
                   const __bf16* __restrict__ wsz, float* __restrict__ y,
                   int nTiles32, int N) {
  __shared__ __align__(16) __bf16 lw[2][16384];   // 2 x 32 KB taps
  const int lane = threadIdx.x & 31;
  const int wave = threadIdx.x >> 5;
  const int t32 = blockIdx.x * 8 + wave;
  const bool valid = t32 < nTiles32;
  const int rowBase = t32 * 32;
  const int r0 = rowBase + (lane & 15);
  const int r1 = r0 + 16;

  v8f zero = {};
  v8f acc0[8], acc1[8];
#pragma unroll
  for (int t = 0; t < 8; ++t) { acc0[t] = zero; acc1[t] = zero; }

#ifdef USE_TDM
  const unsigned ldsBase = (unsigned)(unsigned long long)&lw[0][0];
  if (threadIdx.x < 32) {                  // wave 0 drives the TDM
    tdm_stage(wsz, ldsBase);
    __builtin_amdgcn_s_wait_tensorcnt(0);
  }
  __syncthreads();
#else
  {  // stage tap 0 cooperatively
    const uint4* src = (const uint4*)wsz;
    uint4* dst = (uint4*)&lw[0][0];
    for (int it = threadIdx.x; it < 2048; it += 256) dst[it] = src[it];
  }
  __syncthreads();
#endif

  for (int k = 0; k < 27; ++k) {
    const __bf16* lwbuf = &lw[k & 1][0];

    // kick off next tap's transfer while computing this one
#ifdef USE_TDM
    if (k + 1 < 27 && threadIdx.x < 32)
      tdm_stage(wsz + (long)(k + 1) * 16384, ldsBase + (unsigned)(((k + 1) & 1) * 32768));
#else
    if (k + 1 < 27) {
      const uint4* src = (const uint4*)(wsz + (long)(k + 1) * 16384);
      uint4* dst = (uint4*)&lw[(k + 1) & 1][0];
      for (int it = threadIdx.x; it < 2048; it += 256) dst[it] = src[it];
      __builtin_prefetch(wsz + (long)(k + 2) * 16384 + (long)threadIdx.x * 64, 0, 0);
    }
#endif

    if (valid) {
      int idx0 = (r0 < N) ? nbr[(long)k * N + r0] : -1;
      int idx1 = (r1 < N) ? nbr[(long)k * N + r1] : -1;
      const __bf16* p0 = xb + (long)(idx0 < 0 ? 0 : idx0) * CCH;
      const __bf16* p1 = xb + (long)(idx1 < 0 ? 0 : idx1) * CCH;
#pragma unroll
      for (int kc = 0; kc < 4; ++kc) {
        v16bf a0 = {};
        if (idx0 >= 0) a0 = load_a_frag(p0, kc, lane);
        v16bf a1 = {};
        if (idx1 >= 0) a1 = load_a_frag(p1, kc, lane);
        // batch the 8 B-fragment LDS loads, then issue 16 back-to-back WMMAs
        v16bf bfr[8];
#pragma unroll
        for (int nt = 0; nt < 8; ++nt)
          bfr[nt] = load_b_frag(lwbuf + (kc * 8 + nt) * 512, lane);
        sched_fence();   // keep loads batched ahead of the WMMA burst
#pragma unroll
        for (int nt = 0; nt < 8; ++nt)
          acc0[nt] = wmma_bf16(a0, bfr[nt], acc0[nt]);
#pragma unroll
        for (int nt = 0; nt < 8; ++nt)
          acc1[nt] = wmma_bf16(a1, bfr[nt], acc1[nt]);
      }
    }

    __syncthreads();                       // all readers done with lw[k&1]
#ifdef USE_TDM
    if (k + 1 < 27) {
      if (threadIdx.x < 32) __builtin_amdgcn_s_wait_tensorcnt(0);
      __syncthreads();                     // next buffer ready for everyone
    }
#endif
  }

  if (valid) {
    const int mAdd = (lane >> 4) << 3;
#pragma unroll
    for (int nt = 0; nt < 8; ++nt)
#pragma unroll
      for (int r = 0; r < 8; ++r) {
        int row0 = rowBase + r + mAdd;
        if (row0 < N)      y[(long)row0 * CCH + nt * 16 + (lane & 15)] = acc0[nt][r];
        int row1 = row0 + 16;
        if (row1 < N)      y[(long)row1 * CCH + nt * 16 + (lane & 15)] = acc1[nt][r];
      }
  }
}

// ---------- dense GEMM: Y[N x COUT] = A[N x 128] @ Bswz ----------
template <int COUT>
__global__ __launch_bounds__(256)
void gemm_kernel(const __bf16* __restrict__ A, const __bf16* __restrict__ Bsz,
                 float* __restrict__ Y, int nTiles, int N) {
  const int lane = threadIdx.x & 31;
  const int wave = threadIdx.x >> 5;
  const int tile = blockIdx.x * 8 + wave;
  if (tile >= nTiles) return;
  const int rowBase = tile * 16;
  int aRow = rowBase + (lane & 15);
  if (aRow >= N) aRow = N - 1;
  const __bf16* rowPtr = A + (long)aRow * CCH;
  constexpr int NT = COUT / 16;

  v8f zero = {};
  v8f acc[NT];
#pragma unroll
  for (int t = 0; t < NT; ++t) acc[t] = zero;

  v16bf afr[4];
#pragma unroll
  for (int kc = 0; kc < 4; ++kc) afr[kc] = load_a_frag(rowPtr, kc, lane);

#pragma unroll
  for (int kc = 0; kc < 4; ++kc) {
    v16bf bfr[NT];
#pragma unroll
    for (int nt = 0; nt < NT; ++nt)
      bfr[nt] = load_b_frag(Bsz + (long)(kc * NT + nt) * 512, lane);
    sched_fence();
#pragma unroll
    for (int nt = 0; nt < NT; ++nt)
      acc[nt] = wmma_bf16(afr[kc], bfr[nt], acc[nt]);
  }

  const int mAdd = (lane >> 4) << 3;
#pragma unroll
  for (int nt = 0; nt < NT; ++nt)
#pragma unroll
    for (int r = 0; r < 8; ++r) {
      int row = rowBase + r + mAdd;
      if (row < N) Y[(long)row * COUT + nt * 16 + (lane & 15)] = acc[nt][r];
    }
}

// ---------- BN statistics (two-pass, deterministic) ----------
__global__ __launch_bounds__(256)
void stats_part_kernel(const float* __restrict__ y, float* __restrict__ part,
                       int nCh, int nRows) {
  __shared__ float ssum[256], ssq[256];
  const int t = threadIdx.x;
  const int ch = t & (nCh - 1);
  const int rpi = 256 / nCh;
  const int r0 = t / nCh;
  const int slab = (nRows + gridDim.x - 1) / gridDim.x;
  long start = (long)blockIdx.x * slab;
  long end = start + slab; if (end > nRows) end = nRows;
  float s = 0.f, q = 0.f;
  for (long r = start + r0; r < end; r += rpi) {
    float v = y[r * nCh + ch];
    s += v; q += v * v;
  }
  ssum[t] = s; ssq[t] = q;
  __syncthreads();
  for (int st = 128; st >= nCh; st >>= 1) {
    if (t < st) { ssum[t] += ssum[t + st]; ssq[t] += ssq[t + st]; }
    __syncthreads();
  }
  if (t < nCh) {
    part[(long)blockIdx.x * nCh * 2 + t] = ssum[t];
    part[(long)blockIdx.x * nCh * 2 + nCh + t] = ssq[t];
  }
}

__global__ void stats_final_kernel(const float* __restrict__ part,
                                   const float* __restrict__ g, const float* __restrict__ b,
                                   float* __restrict__ sb, int nCh, int nBlocks, float invN) {
  int c = threadIdx.x;
  if (c >= nCh) return;
  float s = 0.f, q = 0.f;
  for (int i = 0; i < nBlocks; ++i) {
    s += part[(long)i * nCh * 2 + c];
    q += part[(long)i * nCh * 2 + nCh + c];
  }
  float m = s * invN;
  float var = q * invN - m * m;
  float sc = g[c] * rsqrtf(var + 1e-5f);
  sb[c] = sc;
  sb[nCh + c] = b[c] - m * sc;
}

// ---------- BN+ReLU apply ----------
__global__ void act_bf16_kernel(const float* __restrict__ y, const float* __restrict__ sb,
                                __bf16* __restrict__ o, int mask, int nCh, long total) {
  for (long i = (long)blockIdx.x * blockDim.x + threadIdx.x; i < total;
       i += (long)gridDim.x * blockDim.x) {
    int c = (int)(i & mask);
    float v = y[i] * sb[c] + sb[nCh + c];
    o[i] = (__bf16)fmaxf(v, 0.f);
  }
}

__global__ void act_f32_kernel(const float* __restrict__ y, const float* __restrict__ sb,
                               float* __restrict__ o, int mask, int nCh, long total) {
  for (long i = (long)blockIdx.x * blockDim.x + threadIdx.x; i < total;
       i += (long)gridDim.x * blockDim.x) {
    int c = (int)(i & mask);
    float v = y[i] * sb[c] + sb[nCh + c];
    o[i] = fmaxf(v, 0.f);
  }
}

// ---------- expert mix: channelwise convs + pairwise scores + softmax ----------
__global__ __launch_bounds__(128)
void mix_kernel(const float* __restrict__ qact, const __bf16* __restrict__ vact,
                const int* __restrict__ nbr0, const int* __restrict__ nbr1,
                const int* __restrict__ nbr2,
                const float* __restrict__ cb0, const float* __restrict__ cb1,
                const float* __restrict__ cb2,
                __bf16* __restrict__ outb, int N) {
  const int i = blockIdx.x;
  const int c = threadIdx.x;
  __shared__ int idx[41];
  __shared__ float attn[16][3];

  if (c < 7)        idx[c] = nbr0[(long)c * N + i];
  else if (c < 34)  idx[c] = nbr1[(long)(c - 7) * N + i];
  else if (c < 41)  idx[c] = nbr2[(long)(c - 34) * N + i];
  __syncthreads();

  if (c < 16) {
    const int base[4] = {0, 7, 34, 41};
    float qi = qact[(long)i * 16 + c];
    float sc[3];
#pragma unroll
    for (int m = 0; m < 3; ++m) {
      float a = 0.f, cnt = 0.f;
      for (int k = base[m]; k < base[m + 1]; ++k) {
        int id = idx[k];
        if (id >= 0) { a += qact[(long)id * 16 + c]; cnt += 1.f; }
      }
      sc[m] = qi * a / fmaxf(cnt, 1.f);
    }
    float mx = fmaxf(sc[0], fmaxf(sc[1], sc[2]));
    float e0 = __expf(sc[0] - mx), e1 = __expf(sc[1] - mx), e2 = __expf(sc[2] - mx);
    float inv = 1.f / (e0 + e1 + e2);
    attn[c][0] = e0 * inv; attn[c][1] = e1 * inv; attn[c][2] = e2 * inv;
  }
  __syncthreads();

  float o0 = 0.f, o1 = 0.f, o2 = 0.f;
  for (int k = 0; k < 7; ++k) {
    int id = idx[k];
    if (id >= 0) o0 += cb0[k * CCH + c] * (float)vact[(long)id * CCH + c];
  }
  for (int k = 0; k < 27; ++k) {
    int id = idx[7 + k];
    if (id >= 0) o1 += cb1[k * CCH + c] * (float)vact[(long)id * CCH + c];
  }
  for (int k = 0; k < 7; ++k) {
    int id = idx[34 + k];
    if (id >= 0) o2 += cb2[k * CCH + c] * (float)vact[(long)id * CCH + c];
  }
  int v = c >> 3;
  float r = attn[v][0] * o0 + attn[v][1] * o1 + attn[v][2] * o2;
  outb[(long)i * CCH + c] = (__bf16)r;
}

// ---------- final: BN+ReLU + residual ----------
__global__ void final_kernel(const float* __restrict__ y, const float* __restrict__ sb,
                             const float* __restrict__ x, float* __restrict__ out, long total) {
  for (long i = (long)blockIdx.x * blockDim.x + threadIdx.x; i < total;
       i += (long)gridDim.x * blockDim.x) {
    int c = (int)(i & (CCH - 1));
    float v = fmaxf(y[i] * sb[c] + sb[CCH + c], 0.f);
    out[i] = v + x[i];
  }
}

// ---------- driver ----------
extern "C" void kernel_launch(void* const* d_in, const int* in_sizes, int n_in,
                              void* d_out, int out_size, void* d_ws, size_t ws_size,
                              hipStream_t stream) {
  const float* x     = (const float*)d_in[0];
  const float* v1_w  = (const float*)d_in[1];
  const float* v1_g  = (const float*)d_in[2];
  const float* v1_b  = (const float*)d_in[3];
  const float* v2_w  = (const float*)d_in[4];
  const float* v2_g  = (const float*)d_in[5];
  const float* v2_b  = (const float*)d_in[6];
  const float* q_w   = (const float*)d_in[7];
  const float* q_g   = (const float*)d_in[8];
  const float* q_b   = (const float*)d_in[9];
  const float* cb0   = (const float*)d_in[10];
  const float* cb1   = (const float*)d_in[11];
  const float* cb2   = (const float*)d_in[12];
  const float* out_w = (const float*)d_in[13];
  const float* out_g = (const float*)d_in[14];
  const float* out_b = (const float*)d_in[15];
  const int* nbr_cross2 = (const int*)d_in[16];
  const int* nbr_cube   = (const int*)d_in[17];
  const int* nbr_cross3 = (const int*)d_in[18];

  const int N = in_sizes[0] / CCH;
  const int nTiles = (N + 15) / 16;
  const int nTiles32 = (N + 31) / 32;
  const long totNC = (long)N * CCH;
  const long totNQ = (long)N * 16;

  // workspace carve-out
  char* base = (char*)d_ws;
  size_t off = 0;
  auto alloc = [&](size_t bytes) -> void* {
    void* p = base + off;
    off = (off + bytes + 255) & ~(size_t)255;
    return p;
  };
  __bf16* xb    = (__bf16*)alloc(totNC * 2);
  __bf16* vact1 = (__bf16*)alloc(totNC * 2);
  __bf16* vact2 = (__bf16*)alloc(totNC * 2);
  __bf16* ob    = (__bf16*)alloc(totNC * 2);
  float*  ybuf  = (float*)alloc(totNC * 4);
  float*  qraw  = (float*)alloc(totNQ * 4);
  float*  qact  = (float*)alloc(totNQ * 4);
  __bf16* w1s   = (__bf16*)alloc((size_t)27 * CCH * CCH * 2);
  __bf16* w2s   = (__bf16*)alloc((size_t)CCH * CCH * 2);
  __bf16* qws   = (__bf16*)alloc((size_t)CCH * 16 * 2);
  __bf16* ows   = (__bf16*)alloc((size_t)CCH * CCH * 2);
  float*  part  = (float*)alloc((size_t)128 * CCH * 2 * 4);
  float*  sb1   = (float*)alloc(2 * CCH * 4);
  float*  sb2   = (float*)alloc(2 * CCH * 4);
  float*  sb3   = (float*)alloc(2 * 16 * 4);
  float*  sb4   = (float*)alloc(2 * CCH * 4);

  const float invN = 1.0f / (float)N;
  const int gemmGrid = (nTiles + 7) / 8;
  const int spGrid = (nTiles32 + 7) / 8;

  // prep
  cvt_bf16_kernel<<<2048, 256, 0, stream>>>(x, xb, totNC);
  swz_kernel<<<1728, 256, 0, stream>>>(v1_w, w1s, 27, CCH, CCH);
  swz_kernel<<<64, 256, 0, stream>>>(v2_w, w2s, 1, CCH, CCH);
  swz_kernel<<<8, 256, 0, stream>>>(q_w, qws, 1, CCH, 16);
  swz_kernel<<<64, 256, 0, stream>>>(out_w, ows, 1, CCH, CCH);

  // value branch stage 1: 27-tap sparse conv + BN + ReLU
  spconv_kernel<<<spGrid, 256, 0, stream>>>(xb, nbr_cube, w1s, ybuf, nTiles32, N);
  stats_part_kernel<<<128, 256, 0, stream>>>(ybuf, part, CCH, N);
  stats_final_kernel<<<1, 128, 0, stream>>>(part, v1_g, v1_b, sb1, CCH, 128, invN);
  act_bf16_kernel<<<2048, 256, 0, stream>>>(ybuf, sb1, vact1, CCH - 1, CCH, totNC);

  // value branch stage 2: 1x1 conv + BN + ReLU
  gemm_kernel<128><<<gemmGrid, 256, 0, stream>>>(vact1, w2s, ybuf, nTiles, N);
  stats_part_kernel<<<128, 256, 0, stream>>>(ybuf, part, CCH, N);
  stats_final_kernel<<<1, 128, 0, stream>>>(part, v2_g, v2_b, sb2, CCH, 128, invN);
  act_bf16_kernel<<<2048, 256, 0, stream>>>(ybuf, sb2, vact2, CCH - 1, CCH, totNC);

  // query branch: 1x1 conv (Cout=16) + BN + ReLU
  gemm_kernel<16><<<gemmGrid, 256, 0, stream>>>(xb, qws, qraw, nTiles, N);
  stats_part_kernel<<<128, 256, 0, stream>>>(qraw, part, 16, N);
  stats_final_kernel<<<1, 128, 0, stream>>>(part, q_g, q_b, sb3, 16, 128, invN);
  act_f32_kernel<<<512, 256, 0, stream>>>(qraw, sb3, qact, 15, 16, totNQ);

  // experts + attention mix
  mix_kernel<<<N, 128, 0, stream>>>(qact, vact2, nbr_cross2, nbr_cube, nbr_cross3,
                                    cb0, cb1, cb2, ob, N);

  // output projection + BN + ReLU + residual
  gemm_kernel<128><<<gemmGrid, 256, 0, stream>>>(ob, ows, ybuf, nTiles, N);
  stats_part_kernel<<<128, 256, 0, stream>>>(ybuf, part, CCH, N);
  stats_final_kernel<<<1, 128, 0, stream>>>(part, out_g, out_b, sb4, CCH, 128, invN);
  final_kernel<<<2048, 256, 0, stream>>>(ybuf, sb4, x, (float*)d_out, totNC);
}
